// baseline_snn_57208964383542
// MI455X (gfx1250) — matrix operand order repaired and checked by reading, not compiled, not run on hardware
//
#include <hip/hip_runtime.h>
#include <hip/hip_bf16.h>
#include <cstdint>
#include <cstddef>

// ---------------------------------------------------------------------------
// SNN-IIR forward on MI455X (gfx1250, wave32, WMMA).
//  - Time recurrences (dual-exp IIR, LIF): 1 thread per element, loop T=50.
//  - Convs / denses: implicit-GEMM on v_wmma_f32_16x16x32_f16 (f32 accum),
//    weights + im2col offset table staged in LDS, MT accumulators per wave.
//  - Layer 1 conv computed ONCE (input constant over time; IIR is linear).
// ---------------------------------------------------------------------------

typedef __attribute__((ext_vector_type(16))) _Float16 v16h;
typedef __attribute__((ext_vector_type(8)))  _Float16 v8h;
typedef __attribute__((ext_vector_type(8)))  float    v8f;
typedef __attribute__((ext_vector_type(4)))  int      v4i;

#define TT  50
#define BB  16
#define TBB 800   // T*B

// exp(-1/4), exp(-1)
constexpr float SIG = 0.77880078307140487f;
constexpr float ESY = 0.36787944117144233f;
constexpr float A1C = SIG + ESY;          //  1.14668022...
constexpr float A2C = -(SIG * ESY);       // -0.28650480...

// ---------------------------------------------------------------- converters
__global__ void cvt_f16_kernel(const float* __restrict__ in,
                               _Float16* __restrict__ out, int n) {
  int i = blockIdx.x * blockDim.x + threadIdx.x;
  if (i < n) out[i] = (_Float16)in[i];
}

// pad W8 [10,512] -> [16,512] with zero rows (keeps M a multiple of 16)
__global__ void cvt_pad_kernel(const float* __restrict__ in,
                               _Float16* __restrict__ out,
                               int rows, int cols, int padRows) {
  int i = blockIdx.x * blockDim.x + threadIdx.x;
  int n = padRows * cols;
  if (i >= n) return;
  int r = i / cols;
  if (r < rows) out[i] = (_Float16)in[i];
  else          out[i] = (_Float16)0.0f;
}

// ------------------------------------------------------- conv1 (3ch, direct)
__global__ void conv1_kernel(const float* __restrict__ x,
                             const float* __restrict__ w1,
                             float* __restrict__ y1) {
  int idx = blockIdx.x * blockDim.x + threadIdx.x;          // B*32*30*30
  if (idx >= BB * 32 * 30 * 30) return;
  int ow = idx % 30;
  int oh = (idx / 30) % 30;
  int o  = (idx / 900) % 32;
  int b  = idx / (900 * 32);
  const float* xb = x + (size_t)b * 3 * 32 * 32;
  const float* wo = w1 + (size_t)o * 27;
  float acc = 0.f;
#pragma unroll
  for (int ci = 0; ci < 3; ++ci)
#pragma unroll
    for (int kh = 0; kh < 3; ++kh)
#pragma unroll
      for (int kw = 0; kw < 3; ++kw)
        acc += xb[((size_t)ci * 32 + oh + kh) * 32 + ow + kw] * wo[ci * 9 + kh * 3 + kw];
  y1[idx] = acc;
}

// ----------------- layer1: cur[t]=c[t]*conv(x)+b1 -> LIF -> IIR -> f16 (P1)
__global__ void lif1_iir_kernel(const float* __restrict__ y1,
                                const float* __restrict__ b1,
                                _Float16* __restrict__ P1) {
  int idx = blockIdx.x * blockDim.x + threadIdx.x;          // B*32*900
  const int n = BB * 32 * 900;
  if (idx >= n) return;
  int chw = idx % (32 * 900);
  int c   = chw / 900;
  float y = y1[idx], bias = b1[c];
  float c1 = 0.f, c2 = 0.f, v = 0.f, r = 0.f, p1 = 0.f, p2 = 0.f;
  const size_t stride = (size_t)n;                          // B*CHW
  for (int t = 0; t < TT; ++t) {
    float ct = A1C * c1 + A2C * c2 + 1.0f; c2 = c1; c1 = ct;   // IIR of const 1
    float cur = ct * y + bias;
    v = SIG * v + cur - r;
    float s = (v > 1.0f) ? 1.0f : 0.0f;
    r = s;
    float p = A1C * p1 + A2C * p2 + s; p2 = p1; p1 = p;        // IIR of spikes
    P1[(size_t)idx + (size_t)t * stride] = (_Float16)p;
  }
}

// ---------------- generic LIF -> next-layer IIR fusion, f16 output
__global__ void lif_iir_h_kernel(const float* __restrict__ cur,
                                 const float* __restrict__ bias,
                                 int C, int HW,
                                 _Float16* __restrict__ outH) {
  int idx = blockIdx.x * blockDim.x + threadIdx.x;          // B*C*HW
  const int n = BB * C * HW;
  if (idx >= n) return;
  int chw = idx % (C * HW);
  float bs = bias[chw / HW];
  const size_t stride = (size_t)n;
  float v = 0.f, r = 0.f, p1 = 0.f, p2 = 0.f;
  for (int t = 0; t < TT; ++t) {
    float i = cur[(size_t)idx + (size_t)t * stride] + bs;
    v = SIG * v + i - r;
    float s = (v > 1.0f) ? 1.0f : 0.0f;
    r = s;
    float p = A1C * p1 + A2C * p2 + s; p2 = p1; p1 = p;
    outH[(size_t)idx + (size_t)t * stride] = (_Float16)p;
  }
}

// ---------------- generic LIF -> next-layer IIR fusion, f32 output (pre-pool)
__global__ void lif_iir_f_kernel(const float* __restrict__ cur,
                                 const float* __restrict__ bias,
                                 int C, int HW,
                                 float* __restrict__ outF) {
  int idx = blockIdx.x * blockDim.x + threadIdx.x;          // B*C*HW
  const int n = BB * C * HW;
  if (idx >= n) return;
  int chw = idx % (C * HW);
  float bs = bias[chw / HW];
  const size_t stride = (size_t)n;
  float v = 0.f, r = 0.f, p1 = 0.f, p2 = 0.f;
  for (int t = 0; t < TT; ++t) {
    float i = cur[(size_t)idx + (size_t)t * stride] + bs;
    v = SIG * v + i - r;
    float s = (v > 1.0f) ? 1.0f : 0.0f;
    r = s;
    float p = A1C * p1 + A2C * p2 + s; p2 = p1; p1 = p;
    outF[(size_t)idx + (size_t)t * stride] = p;
  }
}

// -------------------------------------------- IIR only (pool output -> f16)
__global__ void iir_kernel(const float* __restrict__ in, int CHW,
                           _Float16* __restrict__ out) {
  int idx = blockIdx.x * blockDim.x + threadIdx.x;          // B*CHW
  const int n = BB * CHW;
  if (idx >= n) return;
  const size_t stride = (size_t)n;
  float p1 = 0.f, p2 = 0.f;
  for (int t = 0; t < TT; ++t) {
    float xv = in[(size_t)idx + (size_t)t * stride];
    float p = A1C * p1 + A2C * p2 + xv; p2 = p1; p1 = p;
    out[(size_t)idx + (size_t)t * stride] = (_Float16)p;
  }
}

// ----------------------------------------------------------- 2x2 max pool
__global__ void maxpool_kernel(const float* __restrict__ in,
                               int C, int H, int W, int OH, int OW,
                               float* __restrict__ out) {
  int idx = blockIdx.x * blockDim.x + threadIdx.x;          // TB*C*OH*OW
  int n = TBB * C * OH * OW;
  if (idx >= n) return;
  int ow = idx % OW;
  int t1 = idx / OW;
  int oh = t1 % OH;
  int t2 = t1 / OH;
  int c  = t2 % C;
  int tb = t2 / C;
  const float* p = in + ((size_t)(tb * C + c) * H + oh * 2) * W + ow * 2;
  out[idx] = fmaxf(fmaxf(p[0], p[1]), fmaxf(p[W], p[W + 1]));
}

// --------------------------- implicit-GEMM conv on WMMA f16 (16x16x32 tiles)
// Out[tb, m, oh, ow] = sum_{ci,kh,kw} Wh[m, ci*9+kh*3+kw] * P[tb, ci, oh+kh, ow+kw]
// Weights + im2col offset table staged in LDS; MT accumulators cover all Cout.
template <int Cin, int Hin, int Win, int Cout, int OH, int OW, int MT>
__global__ void wmma_conv_kernel(const _Float16* __restrict__ Wh,
                                 const _Float16* __restrict__ P,
                                 float* __restrict__ Out,
                                 int nTiles) {
  constexpr int K   = Cin * 9;
  constexpr int OHW = OH * OW;
  constexpr int HW  = Hin * Win;
  __shared__ _Float16 Wlds[Cout * K];
  __shared__ int      offs[K];

  // cooperative stage: weights (b128 copies) + im2col offset table
  {
    const v8h* __restrict__ Wg = (const v8h*)Wh;
    v8h* Wl = (v8h*)Wlds;
    constexpr int NV = (Cout * K) / 8;
    for (int i = threadIdx.x; i < NV; i += 128) Wl[i] = Wg[i];
    for (int k = threadIdx.x; k < K; k += 128) {
      int ci = k / 9;
      int rr = k - ci * 9;
      int kh = rr / 3;
      int kw = rr - kh * 3;
      offs[k] = ci * HW + kh * Win + kw;
    }
  }
  __syncthreads();

  const int wave  = threadIdx.x >> 5;
  const int lane  = threadIdx.x & 31;
  const int tileN = blockIdx.x * 4 + wave;
  if (tileN >= nTiles) return;                 // wave-uniform: EXEC stays all-1

  const int lm  = lane & 15;
  const int lh  = lane >> 4;
  const int n   = tileN * 16 + lm;
  const int tb  = n / OHW;
  const int pos = n - tb * OHW;
  const int oh  = pos / OW;
  const int ow  = pos - oh * OW;
  const _Float16* __restrict__ Pbase = P + ((size_t)tb * Cin * Hin + oh) * Win + ow;

  v8f acc[MT] = {};
#pragma unroll
  for (int k0 = 0; k0 < K; k0 += 32) {
    // B fragment: gather 16 patch values via LDS offset table (aligned b128 reads)
    const int* ob = &offs[k0 + lh * 16];
    v4i o0 = *(const v4i*)(ob + 0);
    v4i o1 = *(const v4i*)(ob + 4);
    v4i o2 = *(const v4i*)(ob + 8);
    v4i o3 = *(const v4i*)(ob + 12);
    v16h bf;
#pragma unroll
    for (int j = 0; j < 4; ++j) bf[j]      = Pbase[o0[j]];
#pragma unroll
    for (int j = 0; j < 4; ++j) bf[4 + j]  = Pbase[o1[j]];
#pragma unroll
    for (int j = 0; j < 4; ++j) bf[8 + j]  = Pbase[o2[j]];
#pragma unroll
    for (int j = 0; j < 4; ++j) bf[12 + j] = Pbase[o3[j]];
    // A fragments from LDS (16B-aligned b128 reads), MT chained WMMAs reuse bf
#pragma unroll
    for (int mt = 0; mt < MT; ++mt) {
      const _Float16* arow = &Wlds[(mt * 16 + lm) * K + k0 + lh * 16];
      v8h lo = *(const v8h*)(arow);
      v8h hi = *(const v8h*)(arow + 8);
      v16h a = __builtin_shufflevector(lo, hi, 0, 1, 2, 3, 4, 5, 6, 7,
                                       8, 9, 10, 11, 12, 13, 14, 15);
      acc[mt] = __builtin_amdgcn_wmma_f32_16x16x32_f16(false, a, false, bf,
                                                       (short)0, acc[mt],
                                                       false, false);
    }
  }
  float* __restrict__ Ob = Out + (size_t)tb * Cout * OHW + pos;
#pragma unroll
  for (int mt = 0; mt < MT; ++mt)
#pragma unroll
    for (int r = 0; r < 8; ++r)
      Ob[(size_t)(mt * 16 + r + lh * 8) * OHW] = acc[mt][r];
}

// ------------------------------------------------- dense GEMM on WMMA f16
// Out[tb, m] = sum_k Wh[m, k] * Q[tb, k]; all loads are aligned b128.
template <int K, int M, int MT>
__global__ void wmma_dense_kernel(const _Float16* __restrict__ Wh,
                                  const _Float16* __restrict__ Q,
                                  float* __restrict__ Out,
                                  int nTiles) {
  const int wave  = threadIdx.x >> 5;
  const int lane  = threadIdx.x & 31;
  const int tileN = blockIdx.x * 4 + wave;
  if (tileN >= nTiles) return;
  const int m0 = blockIdx.y * (16 * MT);
  const int lm = lane & 15;
  const int lh = lane >> 4;
  const int n  = tileN * 16 + lm;
  const _Float16* __restrict__ Brow = Q + (size_t)n * K;
  v8f acc[MT] = {};
  for (int k0 = 0; k0 < K; k0 += 32) {
    v8h blo = *(const v8h*)(Brow + k0 + lh * 16);
    v8h bhi = *(const v8h*)(Brow + k0 + lh * 16 + 8);
    v16h bf = __builtin_shufflevector(blo, bhi, 0, 1, 2, 3, 4, 5, 6, 7,
                                      8, 9, 10, 11, 12, 13, 14, 15);
#pragma unroll
    for (int mt = 0; mt < MT; ++mt) {
      const _Float16* arow = Wh + (size_t)(m0 + mt * 16 + lm) * K + k0 + lh * 16;
      v8h lo = *(const v8h*)(arow);
      v8h hi = *(const v8h*)(arow + 8);
      v16h a = __builtin_shufflevector(lo, hi, 0, 1, 2, 3, 4, 5, 6, 7,
                                       8, 9, 10, 11, 12, 13, 14, 15);
      acc[mt] = __builtin_amdgcn_wmma_f32_16x16x32_f16(false, a, false, bf,
                                                       (short)0, acc[mt],
                                                       false, false);
    }
  }
  float* __restrict__ Ob = Out + (size_t)n * M + m0;
#pragma unroll
  for (int mt = 0; mt < MT; ++mt)
#pragma unroll
    for (int r = 0; r < 8; ++r)
      Ob[mt * 16 + r + lh * 8] = acc[mt][r];
}

// --------------------------------- final LIF; write [B,10,T] spikes to d_out
__global__ void lif8_kernel(const float* __restrict__ C8,
                            const float* __restrict__ b8,
                            float* __restrict__ out) {
  int idx = blockIdx.x * blockDim.x + threadIdx.x;          // B*10
  if (idx >= BB * 10) return;
  int o = idx % 10, b = idx / 10;
  float bias = b8[o];
  float v = 0.f, r = 0.f;
  for (int t = 0; t < TT; ++t) {
    float i = C8[(size_t)(t * BB + b) * 16 + o] + bias;
    v = SIG * v + i - r;
    float s = (v > 1.0f) ? 1.0f : 0.0f;
    r = s;
    out[(size_t)(b * 10 + o) * TT + t] = s;
  }
}

// ===========================================================================
extern "C" void kernel_launch(void* const* d_in, const int* in_sizes, int n_in,
                              void* d_out, int out_size, void* d_ws, size_t ws_size,
                              hipStream_t stream) {
  (void)in_sizes; (void)n_in; (void)out_size; (void)ws_size;
  const float* x  = (const float*)d_in[0];
  const float* w1 = (const float*)d_in[1];
  const float* b1 = (const float*)d_in[2];
  const float* w2 = (const float*)d_in[3];
  const float* b2 = (const float*)d_in[4];
  const float* w3 = (const float*)d_in[5];
  const float* b3 = (const float*)d_in[6];
  const float* w5 = (const float*)d_in[7];
  const float* b5 = (const float*)d_in[8];
  const float* w7 = (const float*)d_in[9];
  const float* b7 = (const float*)d_in[10];
  const float* w8 = (const float*)d_in[11];
  const float* b8 = (const float*)d_in[12];
  float* out = (float*)d_out;

  char* ws = (char*)d_ws;
  size_t off = 0;
  auto alloc = [&](size_t bytes) -> void* {
    void* p = ws + off;
    off += (bytes + 255) & ~(size_t)255;
    return p;
  };

  _Float16* W2H = (_Float16*)alloc(9216ull   * 2);
  _Float16* W3H = (_Float16*)alloc(18432ull  * 2);
  _Float16* W5H = (_Float16*)alloc(36864ull  * 2);
  _Float16* W7H = (_Float16*)alloc(819200ull * 2);
  _Float16* W8H = (_Float16*)alloc(8192ull   * 2);
  float*    Y1  = (float*)   alloc(460800ull * 4);
  _Float16* P1  = (_Float16*)alloc(23040000ull * 2);   // [TB,32,30,30]
  float*    BIGA= (float*)   alloc(34611200ull * 4);   // C2 / C3 / C5 / C7 / C8
  _Float16* P2  = (_Float16*)alloc(20070400ull * 2);   // [TB,32,28,28]
  float*    BIGB= (float*)   alloc(34611200ull * 4);   // R3 / R5 (IIR, pre-pool)
  float*    S4  = (float*)   alloc(8652800ull  * 4);   // [TB,64,13,13]
  _Float16* Q4  = (_Float16*)alloc(8652800ull  * 2);
  float*    S6  = (float*)   alloc(1280000ull  * 4);   // [TB,64,5,5]
  _Float16* Q6  = (_Float16*)alloc(1280000ull  * 2);   // [TB,1600]
  _Float16* Q7  = (_Float16*)alloc(409600ull   * 2);   // [TB,512]

  const int TPB = 256;
  auto nb = [](int n, int t) { return (n + t - 1) / t; };

  // weight conversions (f32 -> f16, row-major [O, Cin*9] / [O, I])
  cvt_f16_kernel<<<nb(9216,  TPB), TPB, 0, stream>>>(w2, W2H, 9216);
  cvt_f16_kernel<<<nb(18432, TPB), TPB, 0, stream>>>(w3, W3H, 18432);
  cvt_f16_kernel<<<nb(36864, TPB), TPB, 0, stream>>>(w5, W5H, 36864);
  cvt_f16_kernel<<<nb(819200,TPB), TPB, 0, stream>>>(w7, W7H, 819200);
  cvt_pad_kernel<<<nb(8192,  TPB), TPB, 0, stream>>>(w8, W8H, 10, 512, 16);

  // layer 1: conv once, then fused c[t]-scale + LIF + IIR
  conv1_kernel   <<<nb(460800, TPB), TPB, 0, stream>>>(x, w1, Y1);
  lif1_iir_kernel<<<nb(460800, TPB), TPB, 0, stream>>>(Y1, b1, P1);

  // layer 2: conv (K=288, N=800*784 -> 39200 tiles), LIF+IIR -> f16
  wmma_conv_kernel<32, 30, 30, 32, 28, 28, 2>
      <<<dim3(9800, 1), 128, 0, stream>>>(W2H, P1, BIGA, 39200);
  lif_iir_h_kernel<<<nb(401408, TPB), TPB, 0, stream>>>(BIGA, b2, 32, 784, P2);

  // layer 3: conv (K=288, N=800*676 -> 33800 tiles), LIF+IIR -> f32
  wmma_conv_kernel<32, 28, 28, 64, 26, 26, 4>
      <<<dim3(8450, 1), 128, 0, stream>>>(W3H, P2, BIGA, 33800);
  lif_iir_f_kernel<<<nb(692224, TPB), TPB, 0, stream>>>(BIGA, b3, 64, 676, BIGB);

  // pool 26->13, then IIR -> f16
  maxpool_kernel<<<nb(8652800, TPB), TPB, 0, stream>>>(BIGB, 64, 26, 26, 13, 13, S4);
  iir_kernel    <<<nb(173056,  TPB), TPB, 0, stream>>>(S4, 64 * 169, Q4);

  // layer 5: conv (K=576, N=800*121 -> 6050 tiles), LIF+IIR -> f32
  wmma_conv_kernel<64, 13, 13, 64, 11, 11, 4>
      <<<dim3(1513, 1), 128, 0, stream>>>(W5H, Q4, BIGA, 6050);
  lif_iir_f_kernel<<<nb(123904, TPB), TPB, 0, stream>>>(BIGA, b5, 64, 121, BIGB);

  // pool 11->5, then IIR -> f16 (flatten [TB,64,5,5] == [TB,1600])
  maxpool_kernel<<<nb(1280000, TPB), TPB, 0, stream>>>(BIGB, 64, 11, 11, 5, 5, S6);
  iir_kernel    <<<nb(25600,   TPB), TPB, 0, stream>>>(S6, 1600, Q6);

  // dense 7: K=1600, M=512, N=800 (50 tiles), MT=4 -> grid.y=8; LIF+IIR -> f16
  wmma_dense_kernel<1600, 512, 4>
      <<<dim3(13, 8), 128, 0, stream>>>(W7H, Q6, BIGA, 50);
  lif_iir_h_kernel<<<nb(8192, TPB), TPB, 0, stream>>>(BIGA, b7, 512, 1, Q7);

  // dense 8: K=512, M=16 (padded), N=800; final LIF writes [B,10,T]
  wmma_dense_kernel<512, 16, 1>
      <<<dim3(13, 1), 128, 0, stream>>>(W8H, Q7, BIGA, 50);
  lif8_kernel<<<1, 256, 0, stream>>>(BIGA, b8, out);
}
